// MLA_67070209294669
// MI455X (gfx1250) — compile-verified, matrix operand order checked by reading, not run on hardware
//
#include <hip/hip_runtime.h>
#include <hip/hip_bf16.h>

// ---------------- problem constants ----------------
#define DIM      2048
#define N_HEADS  16
#define KV_LORA  512
#define QK_NOPE  128
#define QK_ROPE  64
#define V_DIM    128
#define QK_HEAD  192          // QK_NOPE + QK_ROPE
#define BB       2
#define SS       2048
#define MTOT     (BB*SS)      // 4096 tokens
#define QEXT     576          // KV_LORA + QK_ROPE
#define LDSPAD   2056         // f32 score row stride in LDS
#define PSTR     2080         // bf16 prob row stride in LDS (16B-aligned rows)

typedef unsigned short bf16_t;
typedef __attribute__((ext_vector_type(16))) __bf16 v16bf;
typedef __attribute__((ext_vector_type(8)))  float  v8f;

// ---------------- scalar conversions ----------------
__device__ __forceinline__ bf16_t f2bf(float f) {
  __hip_bfloat16 h = __float2bfloat16(f);          // native cvt on gfx1250
  return *reinterpret_cast<bf16_t*>(&h);
}
__device__ __forceinline__ float bf2f(bf16_t u) {
  return __uint_as_float(((unsigned int)u) << 16);
}

__device__ __forceinline__ v8f vzero() {
  v8f z;
#pragma unroll
  for (int i = 0; i < 8; i++) z[i] = 0.0f;
  return z;
}

// ---------------- bf16 fragment loader ----------------
// 16(row) x 32(k) bf16 fragment from row-major bf16 storage (ld in elements,
// ld % 8 == 0, base 16B-aligned). gfx1250 16-bit A/B layout: lane&15 = row,
// kh = (lane>>4)*8; VGPRs 0-3 hold k = kh..kh+7, VGPRs 4-7 hold k = 16+kh..+7.
// => exactly two contiguous 16-byte loads per lane, zero VALU.
__device__ __forceinline__ v16bf frag_ld(const bf16_t* __restrict__ base, int ld, int lane) {
  int r  = lane & 15;
  int kh = (lane >> 4) << 3;
  const bf16_t* p = base + (size_t)r * ld + kh;
  union { v16bf v; uint4 q[2]; } f;
  f.q[0] = *(const uint4*)(p);
  f.q[1] = *(const uint4*)(p + 16);
  return f.v;
}

__device__ __forceinline__ v8f wmma_bf16(v16bf a, v16bf b, v8f c) {
  return __builtin_amdgcn_wmma_f32_16x16x32_bf16(false, a, false, b, (short)0, c, false, false);
}

// C/D layout: n = lane&15, m = r + 8*(lane>>4)
__device__ __forceinline__ void store_tile_f32(float* base, int ld, int lane, v8f c, float bv) {
  int n  = lane & 15;
  int mh = (lane >> 4) * 8;
#pragma unroll
  for (int r = 0; r < 8; r++) base[(size_t)(mh + r) * ld + n] = c[r] + bv;
}
__device__ __forceinline__ void store_tile_bf(bf16_t* base, int ld, int lane, v8f c, float bv) {
  int n  = lane & 15;
  int mh = (lane >> 4) * 8;
#pragma unroll
  for (int r = 0; r < 8; r++) base[(size_t)(mh + r) * ld + n] = f2bf(c[r] + bv);
}

// ---------------- conversion kernels ----------------
__global__ void cvt_f32_to_bf16(const float* __restrict__ in, bf16_t* __restrict__ out, long n) {
  long i = (long)blockIdx.x * blockDim.x + threadIdx.x;
  long stride = (long)gridDim.x * blockDim.x;
  for (; i * 2 + 1 < n; i += stride) {
    float2 v = ((const float2*)in)[i];
    unsigned int pk = (unsigned int)f2bf(v.x) | ((unsigned int)f2bf(v.y) << 16);
    ((unsigned int*)out)[i] = pk;
  }
}

// wb1_t[h][n][k] = wkv_b[h][k][n]   (absorb half, transposed -> row-major contiguous)
// wb2  [h][d][c] = wkv_b[h][128+d][c]
__global__ void prep_wb(const float* __restrict__ wkv_b,
                        bf16_t* __restrict__ wb1_t, bf16_t* __restrict__ wb2) {
  long stride = (long)gridDim.x * blockDim.x;
  long i = (long)blockIdx.x * blockDim.x + threadIdx.x;
  const long n1 = (long)N_HEADS * KV_LORA * QK_NOPE;
  for (long t = i; t < n1; t += stride) {
    long h = t / ((long)KV_LORA * QK_NOPE);
    long rem = t % ((long)KV_LORA * QK_NOPE);
    long n = rem / QK_NOPE, k = rem % QK_NOPE;
    wb1_t[t] = f2bf(wkv_b[(h * 2 * QK_NOPE + k) * KV_LORA + n]);
  }
  const long n2 = (long)N_HEADS * V_DIM * KV_LORA;
  for (long t = i; t < n2; t += stride) {
    long h = t / ((long)V_DIM * KV_LORA);
    long rem = t % ((long)V_DIM * KV_LORA);
    long d = rem / KV_LORA, c = rem % KV_LORA;
    wb2[t] = f2bf(wkv_b[(h * 2 * QK_NOPE + QK_NOPE + d) * KV_LORA + c]);
  }
}

// ---------------- GEMM: C[M,N] = A[M,K] @ W[N,K]^T + bias ----------------
// block = 256 (8 waves = 4m x 2n); wave = 32x64 macro-tile; block tile = 128x128.
// Per K-step: 2 A-frags + 4 B-frags (12 b128 loads) feed 8 WMMAs.
template <bool OUT_BF16>
__global__ void gemm_nt_bias(const bf16_t* __restrict__ A, int lda,
                             const bf16_t* __restrict__ W,   // [N,K] row-major
                             const float* __restrict__ bias,
                             void* __restrict__ Cv, int ldc,
                             int N, int K) {
  const int lane = threadIdx.x & 31, wave = threadIdx.x >> 5;
  const int wm = wave >> 1, wn = wave & 1;
  const int m0 = blockIdx.x * 128 + wm * 32;
  const int nb = blockIdx.y * 128 + wn * 64;
  bool ok[4];
#pragma unroll
  for (int j = 0; j < 4; j++) ok[j] = (nb + 16 * j) < N;

  v8f acc[2][4];
#pragma unroll
  for (int i = 0; i < 2; i++)
#pragma unroll
    for (int j = 0; j < 4; j++) acc[i][j] = vzero();

  const bf16_t* A0 = A + (size_t)m0 * lda;
  const bf16_t* A1 = A + (size_t)(m0 + 16) * lda;
  for (int kc = 0; kc < K; kc += 32) {
    v16bf fa0 = frag_ld(A0 + kc, lda, lane);
    v16bf fa1 = frag_ld(A1 + kc, lda, lane);
#pragma unroll
    for (int j = 0; j < 4; j++) {
      if (ok[j]) {
        v16bf fb = frag_ld(W + (size_t)(nb + 16 * j) * K + kc, K, lane);
        acc[0][j] = wmma_bf16(fa0, fb, acc[0][j]);
        acc[1][j] = wmma_bf16(fa1, fb, acc[1][j]);
      }
    }
  }
#pragma unroll
  for (int j = 0; j < 4; j++) {
    if (!ok[j]) continue;
    int n0 = nb + 16 * j;
    float bv = bias[n0 + (lane & 15)];
    if (OUT_BF16) {
      bf16_t* C = (bf16_t*)Cv;
      store_tile_bf(C + (size_t)m0 * ldc + n0, ldc, lane, acc[0][j], bv);
      store_tile_bf(C + (size_t)(m0 + 16) * ldc + n0, ldc, lane, acc[1][j], bv);
    } else {
      float* C = (float*)Cv;
      store_tile_f32(C + (size_t)m0 * ldc + n0, ldc, lane, acc[0][j], bv);
      store_tile_f32(C + (size_t)(m0 + 16) * ldc + n0, ldc, lane, acc[1][j], bv);
    }
  }
}

// ---------------- prep: rms-norm + rope ----------------
__global__ void prep_kernel(const float* __restrict__ kv_full, // [M,576] f32
                            const bf16_t* __restrict__ q,      // [M,3072] bf16
                            const float* __restrict__ kv_w,    // [512]
                            bf16_t* __restrict__ k_ext,        // [M,576]
                            bf16_t* __restrict__ v_t,          // [B][512][S]
                            bf16_t* __restrict__ q_ext) {      // [M,16,576]
  const int m = blockIdx.x, tid = threadIdx.x;
  const int s = m % SS, b = m / SS;
  __shared__ float red[256];
  __shared__ float rms_s;

  const float* kvrow = kv_full + (size_t)m * QEXT;
  float acc = 0.f;
  for (int i = tid; i < KV_LORA; i += 256) { float v = kvrow[i]; acc += v * v; }
  red[tid] = acc;
  __syncthreads();
  for (int off = 128; off > 0; off >>= 1) {
    if (tid < off) red[tid] += red[tid + off];
    __syncthreads();
  }
  if (tid == 0) rms_s = rsqrtf(red[0] / (float)KV_LORA + 1e-6f);
  __syncthreads();
  const float rms = rms_s;

  bf16_t* krow = k_ext + (size_t)m * QEXT;
  bf16_t* vcol = v_t + (size_t)b * KV_LORA * SS + s;
  for (int i = tid; i < KV_LORA; i += 256) {
    bf16_t v = f2bf(kvrow[i] * rms * kv_w[i]);
    krow[i] = v;
    vcol[(size_t)i * SS] = v;
  }

  if (tid < 32) {                                   // k_pe rope, d = 64
    int i = tid;
    float inv = powf(10000.0f, -(2.0f * i) / 64.0f);
    float ang = (float)s * inv;
    float c = __cosf(ang), sn = __sinf(ang);
    float x1 = kvrow[KV_LORA + 2 * i], x2 = kvrow[KV_LORA + 2 * i + 1];
    krow[KV_LORA + 2 * i]     = f2bf(x1 * c - x2 * sn);
    krow[KV_LORA + 2 * i + 1] = f2bf(x1 * sn + x2 * c);
  }

  const bf16_t* qrow = q + (size_t)m * (N_HEADS * QK_HEAD);
  bf16_t* qe = q_ext + (size_t)m * N_HEADS * QEXT;
  for (int i = tid; i < 512; i += 256) {            // 512 pairs over flat 1024
    int j0 = 2 * i;                                  // pair stays inside a head
    int h = j0 >> 6, r = j0 & 63;
    float inv = powf(10000.0f, -(2.0f * i) / 1024.0f);
    float ang = (float)s * inv;
    float c = __cosf(ang), sn = __sinf(ang);
    float x1 = bf2f(qrow[h * QK_HEAD + QK_NOPE + r]);
    float x2 = bf2f(qrow[h * QK_HEAD + QK_NOPE + r + 1]);
    qe[(size_t)h * QEXT + KV_LORA + r]     = f2bf(x1 * c - x2 * sn);
    qe[(size_t)h * QEXT + KV_LORA + r + 1] = f2bf(x1 * sn + x2 * c);
  }
}

// ---------------- absorb: q_ext[m,h,0:512] = q_nope @ wb1 ----------------
// wave 32x64; block 128x128; N = 512 (grid.y = 4, no guards), K = 128.
__global__ void absorb_q(const bf16_t* __restrict__ q,
                         const bf16_t* __restrict__ wb1_t,   // [16][512][128]
                         bf16_t* __restrict__ q_ext) {
  const int lane = threadIdx.x & 31, wave = threadIdx.x >> 5;
  const int wm = wave >> 1, wn = wave & 1;
  const int h  = blockIdx.z;
  const int m0 = blockIdx.x * 128 + wm * 32;
  const int nb = blockIdx.y * 128 + wn * 64;

  const bf16_t* A0 = q + (size_t)m0 * (N_HEADS * QK_HEAD) + h * QK_HEAD;
  const bf16_t* A1 = A0 + (size_t)16 * (N_HEADS * QK_HEAD);
  const bf16_t* Wh = wb1_t + (size_t)h * KV_LORA * QK_NOPE;

  v8f acc[2][4];
#pragma unroll
  for (int i = 0; i < 2; i++)
#pragma unroll
    for (int j = 0; j < 4; j++) acc[i][j] = vzero();

  for (int kc = 0; kc < QK_NOPE; kc += 32) {
    v16bf fa0 = frag_ld(A0 + kc, N_HEADS * QK_HEAD, lane);
    v16bf fa1 = frag_ld(A1 + kc, N_HEADS * QK_HEAD, lane);
#pragma unroll
    for (int j = 0; j < 4; j++) {
      v16bf fb = frag_ld(Wh + (size_t)(nb + 16 * j) * QK_NOPE + kc, QK_NOPE, lane);
      acc[0][j] = wmma_bf16(fa0, fb, acc[0][j]);
      acc[1][j] = wmma_bf16(fa1, fb, acc[1][j]);
    }
  }
  bf16_t* Cb = q_ext + ((size_t)m0 * N_HEADS + h) * QEXT;
  const int ldc = N_HEADS * QEXT;
#pragma unroll
  for (int j = 0; j < 4; j++) {
    store_tile_bf(Cb + nb + 16 * j, ldc, lane, acc[0][j], 0.f);
    store_tile_bf(Cb + (size_t)16 * ldc + nb + 16 * j, ldc, lane, acc[1][j], 0.f);
  }
}

// ---------------- causal attention: one (b,h,16-query block) per WG ----------
__device__ __forceinline__ void write_scores(float* sc, v8f acc, int t0, int s0,
                                             int lane, float scale) {
  int t  = t0 + (lane & 15);
  int mh = (lane >> 4) * 8;
#pragma unroll
  for (int r = 0; r < 8; r++) {
    int row = mh + r;
    float v = acc[r] * scale;
    if (t > s0 + row) v = -1e30f;         // causal mask
    sc[(size_t)row * LDSPAD + t] = v;
  }
}

__global__ void attn_kernel(const bf16_t* __restrict__ q_ext,  // [M,16,576]
                            const bf16_t* __restrict__ k_ext,  // [M,576]
                            const bf16_t* __restrict__ v_t,    // [B][512][S]
                            bf16_t* __restrict__ out_lat) {    // [M,16,512]
  const int lane = threadIdx.x & 31, wave = threadIdx.x >> 5, tid = threadIdx.x;
  const int s0 = blockIdx.x * 16;
  const int h  = blockIdx.y;
  const int b  = blockIdx.z;

  extern __shared__ char smem_raw[];
  float*  sc = (float*)smem_raw;                          // 16 x LDSPAD f32
  bf16_t* pb = (bf16_t*)(smem_raw + 16 * LDSPAD * 4);     // 16 x PSTR bf16
  bf16_t* qs = pb + 16 * PSTR;                            // 16 x 576 bf16

  const int ncols     = s0 + 16;                          // causal extent
  const int ncols_pad = (ncols + 31) & ~31;
  const float scale   = rsqrtf((float)QK_HEAD);

  for (int idx = tid; idx < 16 * QEXT; idx += 256) {      // stage Q block
    int r = idx / QEXT, c = idx % QEXT;
    qs[idx] = q_ext[(((size_t)(b * SS + s0 + r)) * N_HEADS + h) * QEXT + c];
  }
  __syncthreads();

  const bf16_t* Kb = k_ext + (size_t)b * SS * QEXT;

  // pass 1: scores = Q @ K^T; two K-tiles per iteration share one Q fragment
  const int ntt = ncols >> 4;
  for (int tp = wave; tp * 2 < ntt; tp += 8) {
    int t0 = tp * 32;
    bool have2 = (tp * 2 + 1) < ntt;      // uniform per wave
    v8f acc0 = vzero(), acc1 = vzero();
    for (int kc = 0; kc < QEXT; kc += 32) {
      v16bf fa  = frag_ld(qs + kc, QEXT, lane);
      v16bf fb0 = frag_ld(Kb + (size_t)t0 * QEXT + kc, QEXT, lane);
      acc0 = wmma_bf16(fa, fb0, acc0);
      if (have2) {
        v16bf fb1 = frag_ld(Kb + (size_t)(t0 + 16) * QEXT + kc, QEXT, lane);
        acc1 = wmma_bf16(fa, fb1, acc1);
      }
    }
    write_scores(sc, acc0, t0, s0, lane, scale);
    if (have2) write_scores(sc, acc1, t0 + 16, s0, lane, scale);
  }
  // zero-probability padding for the ragged 16 columns
  if (ncols_pad > ncols) {
    int w = ncols_pad - ncols;
    for (int idx = tid; idx < 16 * w; idx += 256) {
      int r = idx / w, c = ncols + idx % w;
      pb[(size_t)r * PSTR + c] = 0;
    }
  }
  __syncthreads();

  // softmax: wave handles rows 2*wave, 2*wave+1; writes bf16 probs once
  for (int rr = 0; rr < 2; rr++) {
    int row = wave * 2 + rr;
    float* prow = sc + (size_t)row * LDSPAD;
    bf16_t* pbr = pb + (size_t)row * PSTR;
    float mx = -3.0e38f;
    for (int t = lane; t < ncols; t += 32) mx = fmaxf(mx, prow[t]);
#pragma unroll
    for (int off = 16; off > 0; off >>= 1) mx = fmaxf(mx, __shfl_xor(mx, off, 32));
    float sum = 0.f;
    for (int t = lane; t < ncols; t += 32) {
      float p = __expf(prow[t] - mx);
      prow[t] = p;
      sum += p;
    }
#pragma unroll
    for (int off = 16; off > 0; off >>= 1) sum += __shfl_xor(sum, off, 32);
    float inv = 1.0f / sum;
    for (int t = lane; t < ncols; t += 32) pbr[t] = f2bf(prow[t] * inv);
  }
  __syncthreads();

  // pass 2: out = P @ V using transposed V (rows = latent dim, ld = S)
  const int n0 = wave * 64;               // 8 waves cover N = 512
  const bf16_t* Vb = v_t + (size_t)b * KV_LORA * SS;
  v8f acc2[4];
#pragma unroll
  for (int j = 0; j < 4; j++) acc2[j] = vzero();
  for (int kc = 0; kc < ncols_pad; kc += 32) {
    v16bf fa = frag_ld(pb + kc, PSTR, lane);
#pragma unroll
    for (int j = 0; j < 4; j++) {
      v16bf fb = frag_ld(Vb + (size_t)(n0 + 16 * j) * SS + kc, SS, lane);
      acc2[j] = wmma_bf16(fa, fb, acc2[j]);
    }
  }
  bf16_t* obase = out_lat + (((size_t)(b * SS + s0)) * N_HEADS + h) * KV_LORA;
  const int ldo = N_HEADS * KV_LORA;
#pragma unroll
  for (int j = 0; j < 4; j++) store_tile_bf(obase + n0 + 16 * j, ldo, lane, acc2[j], 0.f);
}

// ---------------- per-head V projection ----------------
// wave 32x64; N = 128 covered by 2 n-waves; K = 512.
__global__ void vproj_kernel(const bf16_t* __restrict__ out_lat, // [M,16,512]
                             const bf16_t* __restrict__ wb2,     // [16][128][512]
                             bf16_t* __restrict__ out_v) {       // [M,2048]
  const int lane = threadIdx.x & 31, wave = threadIdx.x >> 5;
  const int wm = wave >> 1, wn = wave & 1;
  const int h  = blockIdx.z;
  const int m0 = blockIdx.x * 128 + wm * 32;
  const int nb = wn * 64;

  const bf16_t* A0 = out_lat + ((size_t)m0 * N_HEADS + h) * KV_LORA;
  const bf16_t* A1 = A0 + (size_t)16 * N_HEADS * KV_LORA;
  const int lda    = N_HEADS * KV_LORA;
  const bf16_t* Wh = wb2 + (size_t)h * V_DIM * KV_LORA;

  v8f acc[2][4];
#pragma unroll
  for (int i = 0; i < 2; i++)
#pragma unroll
    for (int j = 0; j < 4; j++) acc[i][j] = vzero();

  for (int kc = 0; kc < KV_LORA; kc += 32) {
    v16bf fa0 = frag_ld(A0 + kc, lda, lane);
    v16bf fa1 = frag_ld(A1 + kc, lda, lane);
#pragma unroll
    for (int j = 0; j < 4; j++) {
      v16bf fb = frag_ld(Wh + (size_t)(nb + 16 * j) * KV_LORA + kc, KV_LORA, lane);
      acc[0][j] = wmma_bf16(fa0, fb, acc[0][j]);
      acc[1][j] = wmma_bf16(fa1, fb, acc[1][j]);
    }
  }
  bf16_t* Cb = out_v + (size_t)m0 * DIM + h * V_DIM;
#pragma unroll
  for (int j = 0; j < 4; j++) {
    store_tile_bf(Cb + nb + 16 * j, DIM, lane, acc[0][j], 0.f);
    store_tile_bf(Cb + (size_t)16 * DIM + nb + 16 * j, DIM, lane, acc[1][j], 0.f);
  }
}

// ---------------- launcher ----------------
extern "C" void kernel_launch(void* const* d_in, const int* in_sizes, int n_in,
                              void* d_out, int out_size, void* d_ws, size_t ws_size,
                              hipStream_t stream) {
  (void)in_sizes; (void)n_in; (void)out_size; (void)ws_size;
  const float* x     = (const float*)d_in[0];
  // d_in[1] = mask (causal; implemented directly in attn_kernel)
  const float* wq    = (const float*)d_in[2];
  const float* bq    = (const float*)d_in[3];
  const float* wkv_a = (const float*)d_in[4];
  const float* bkv_a = (const float*)d_in[5];
  const float* kv_w  = (const float*)d_in[6];
  const float* wkv_b = (const float*)d_in[7];
  const float* wo    = (const float*)d_in[8];
  const float* bo    = (const float*)d_in[9];
  float* out = (float*)d_out;

  char* ws = (char*)d_ws;
  size_t o = 0;
  auto carve = [&](size_t bytes) -> char* {
    char* p = ws + o;
    o += (bytes + 255) & ~(size_t)255;
    return p;
  };
  bf16_t* x_bf     = (bf16_t*)carve((size_t)MTOT * DIM * 2);
  bf16_t* wq_bf    = (bf16_t*)carve((size_t)N_HEADS * QK_HEAD * DIM * 2);
  bf16_t* wkv_a_bf = (bf16_t*)carve((size_t)QEXT * DIM * 2);
  bf16_t* wo_bf    = (bf16_t*)carve((size_t)DIM * DIM * 2);
  bf16_t* wb1_t    = (bf16_t*)carve((size_t)N_HEADS * KV_LORA * QK_NOPE * 2);
  bf16_t* wb2      = (bf16_t*)carve((size_t)N_HEADS * V_DIM * KV_LORA * 2);
  bf16_t* q_bf     = (bf16_t*)carve((size_t)MTOT * N_HEADS * QK_HEAD * 2);
  float*  kvfull   = (float*)carve((size_t)MTOT * QEXT * 4);
  bf16_t* q_ext    = (bf16_t*)carve((size_t)MTOT * N_HEADS * QEXT * 2);
  bf16_t* k_ext    = (bf16_t*)carve((size_t)MTOT * QEXT * 2);
  bf16_t* v_t      = (bf16_t*)carve((size_t)BB * KV_LORA * SS * 2);
  bf16_t* out_lat  = (bf16_t*)carve((size_t)MTOT * N_HEADS * KV_LORA * 2);
  bf16_t* out_v    = (bf16_t*)carve((size_t)MTOT * DIM * 2);

  dim3 blk(256);

  // one-shot bf16 conversions
  cvt_f32_to_bf16<<<dim3(2048), blk, 0, stream>>>(x, x_bf, (long)MTOT * DIM);
  cvt_f32_to_bf16<<<dim3(2048), blk, 0, stream>>>(wq, wq_bf, (long)N_HEADS * QK_HEAD * DIM);
  cvt_f32_to_bf16<<<dim3(512), blk, 0, stream>>>(wkv_a, wkv_a_bf, (long)QEXT * DIM);
  cvt_f32_to_bf16<<<dim3(2048), blk, 0, stream>>>(wo, wo_bf, (long)DIM * DIM);
  prep_wb<<<dim3(1024), blk, 0, stream>>>(wkv_b, wb1_t, wb2);

  // q = x @ wq^T + bq  -> bf16
  gemm_nt_bias<true><<<dim3(MTOT / 128, (N_HEADS * QK_HEAD) / 128), blk, 0, stream>>>(
      x_bf, DIM, wq_bf, bq, q_bf, N_HEADS * QK_HEAD, N_HEADS * QK_HEAD, DIM);

  // kv_full = x @ wkv_a^T + bkv_a -> f32 (rms-norm precision)
  gemm_nt_bias<false><<<dim3(MTOT / 128, (QEXT + 127) / 128), blk, 0, stream>>>(
      x_bf, DIM, wkv_a_bf, bkv_a, kvfull, QEXT, QEXT, DIM);

  // rmsnorm + rope -> k_ext (bf16), v_t (bf16, transposed), q_ext pe cols
  prep_kernel<<<dim3(MTOT), blk, 0, stream>>>(kvfull, q_bf, kv_w, k_ext, v_t, q_ext);

  // q_nope absorb into latent space -> q_ext[:, :, :512]
  absorb_q<<<dim3(MTOT / 128, KV_LORA / 128, N_HEADS), blk, 0, stream>>>(q_bf, wb1_t, q_ext);

  // causal attention
  size_t shmem = (size_t)16 * LDSPAD * 4 + (size_t)16 * PSTR * 2 + (size_t)16 * QEXT * 2;
  attn_kernel<<<dim3(SS / 16, N_HEADS, BB), blk, shmem, stream>>>(q_ext, k_ext, v_t, out_lat);

  // per-head V projection -> bf16
  vproj_kernel<<<dim3(MTOT / 128, 1, N_HEADS), blk, 0, stream>>>(out_lat, wb2, out_v);

  // final: out = out_v @ wo^T + bo -> f32 output
  gemm_nt_bias<false><<<dim3(MTOT / 128, DIM / 128), blk, 0, stream>>>(
      out_v, DIM, wo_bf, bo, out, DIM, DIM, DIM);
}
